// EGNNModel_70025146794720
// MI455X (gfx1250) — matrix operand order repaired and checked by reading, not compiled, not run on hardware
//
#include <hip/hip_runtime.h>

#define N_NODES 100000
#define N_EDGES 1600000
#define FDIM 64
#define DEPTH 4
#define NWAVES 8

typedef __attribute__((ext_vector_type(16))) __bf16 bf16x16;
typedef __attribute__((ext_vector_type(8)))  __bf16 bf16x8;
typedef __attribute__((ext_vector_type(8)))  float  f32x8;

static __device__ __forceinline__ float silu_f(float z) {
    return z * (1.0f / (1.0f + __expf(-z)));
}

// A fragment for V_WMMA_F32_16X16X32_BF16 from row-major LDS tile [16][stride].
// 16-bit A 16x32 layout: lane L -> row L&15; lanes 0-15 hold K = k0+{0..7, 16..23},
// lanes 16-31 hold K = k0+{8..15, 24..31}.
static __device__ __forceinline__ bf16x16 make_a_frag(const __bf16* base, int stride,
                                                      int k0, int lane) {
    const int row = lane & 15;
    const int hi  = (lane >> 4) & 1;
    const __bf16* p = base + row * stride + k0 + hi * 8;
    bf16x8 lo = *(const bf16x8*)(p);
    bf16x8 hb = *(const bf16x8*)(p + 16);
    bf16x16 a;
#pragma unroll
    for (int i = 0; i < 8; ++i) { a[i] = lo[i]; a[i + 8] = hb[i]; }
    return a;
}

// B fragment (32x16) from transposed weights wT[n][k] row-major with row stride kstride.
// Lane L -> column n0 + (L&15); lanes 0-15 hold K = k0+{0..7, 16..23}, lanes 16-31 K = k0+{8..15, 24..31}.
static __device__ __forceinline__ bf16x16 make_b_frag(const __bf16* wT, int kstride,
                                                      int n0, int k0, int lane) {
    const int col = lane & 15;
    const int hi  = (lane >> 4) & 1;
    const __bf16* p = wT + (n0 + col) * kstride + k0 + hi * 8;
    bf16x8 lo = *(const bf16x8*)(p);
    bf16x8 hb = *(const bf16x8*)(p + 16);
    bf16x16 b;
#pragma unroll
    for (int i = 0; i < 8; ++i) { b[i] = lo[i]; b[i + 8] = hb[i]; }
    return b;
}

static __device__ __forceinline__ f32x8 wmma_bf16(bf16x16 a, bf16x16 b, f32x8 c) {
    return __builtin_amdgcn_wmma_f32_16x16x32_bf16(false, a, false, b, (short)0, c,
                                                   false, false);
}

static __device__ __forceinline__ void fence_sched() { asm volatile("" ::: "memory"); }

// ---------------------------------------------------------------------------
// Edge kernel: per 16-edge tile per wave:
//   f = [h[src] | h[dst] | radial] -> m = silu(silu(f@ew1+eb1)@ew2+eb2)
//   c = silu(m@cw1+cb1)@cw2 ;  atomics: hne[dst]+=m, xne[dst]+=c*xdiff_norm
// ---------------------------------------------------------------------------
__global__ __launch_bounds__(256) void egnn_edge_kernel(
    const float* __restrict__ h, const float* __restrict__ x,
    const int* __restrict__ src, const int* __restrict__ dst,
    const float* __restrict__ ew1, const float* __restrict__ eb1,
    const float* __restrict__ ew2, const float* __restrict__ eb2,
    const float* __restrict__ cw1, const float* __restrict__ cb1,
    const float* __restrict__ cw2,
    float* __restrict__ hne, float* __restrict__ xne)
{
    __shared__ __bf16 sE1[64 * 128];          // ew1[0:128]^T : [n][k]
    __shared__ __bf16 sE2[64 * 64];           // ew2^T
    __shared__ __bf16 sC1[64 * 64];           // cw1^T
    __shared__ __bf16 sA[NWAVES][16 * 128];   // per-wave A staging (edges x 2F)
    __shared__ __bf16 sM[NWAVES][16 * 64];    // per-wave intermediate staging
    __shared__ float  sRad[NWAVES][16];
    __shared__ float  sXdn[NWAVES][16][3];
    __shared__ int    sDst[NWAVES][16];

    const int tid = threadIdx.x;
    for (int i = tid; i < 64 * 128; i += 256) {
        int n = i >> 7, k = i & 127;
        sE1[i] = (__bf16)ew1[k * 64 + n];
    }
    for (int i = tid; i < 64 * 64; i += 256) {
        int n = i >> 6, k = i & 63;
        sE2[i] = (__bf16)ew2[k * 64 + n];
        sC1[i] = (__bf16)cw1[k * 64 + n];
    }
    __syncthreads();

    const int lane = tid & 31;
    const int wv   = tid >> 5;
    const int col  = lane & 15;
    const int hi   = (lane >> 4) & 1;

    float eb1r[4], eb2r[4], cb1r[4], wradr[4], cw2r[4];
#pragma unroll
    for (int n = 0; n < 4; ++n) {
        eb1r[n]  = eb1[n * 16 + col];
        eb2r[n]  = eb2[n * 16 + col];
        cb1r[n]  = cb1[n * 16 + col];
        wradr[n] = ew1[128 * 64 + n * 16 + col];   // radial row of ew1
        cw2r[n]  = cw2[n * 16 + col];
    }

    __bf16* Ab = sA[wv];
    __bf16* Mb = sM[wv];

    const int ntiles = (N_EDGES + 15) >> 4;
    const int nwtot  = gridDim.x * NWAVES;

    for (int tile = blockIdx.x * NWAVES + wv; tile < ntiles; tile += nwtot) {
        const int e0   = tile << 4;
        const int erow = lane & 15;
        const int e    = min(e0 + erow, N_EDGES - 1);
        const int cnt  = min(16, N_EDGES - e0);

        const int sidx = src[e];
        const int didx = dst[e];
        const int node = hi ? didx : sidx;

        if (!hi) {
            float dx = x[sidx * 3 + 0] - x[didx * 3 + 0];
            float dy = x[sidx * 3 + 1] - x[didx * 3 + 1];
            float dz = x[sidx * 3 + 2] - x[didx * 3 + 2];
            float rad = dx * dx + dy * dy + dz * dz;
            float inv = 1.0f / (sqrtf(rad) + 1e-30f);
            sRad[wv][erow]    = rad;
            sXdn[wv][erow][0] = dx * inv;
            sXdn[wv][erow][1] = dy * inv;
            sXdn[wv][erow][2] = dz * inv;
            sDst[wv][erow]    = didx;
        }

        // stage h[src] (lanes 0-15) / h[dst] (lanes 16-31) as bf16 rows
        {
            const float* hp = h + (long)node * 64;
#pragma unroll
            for (int f0 = 0; f0 < 64; f0 += 8) {
                float4 a0 = *(const float4*)(hp + f0);
                float4 a1 = *(const float4*)(hp + f0 + 4);
                bf16x8 vb;
                vb[0] = (__bf16)a0.x; vb[1] = (__bf16)a0.y;
                vb[2] = (__bf16)a0.z; vb[3] = (__bf16)a0.w;
                vb[4] = (__bf16)a1.x; vb[5] = (__bf16)a1.y;
                vb[6] = (__bf16)a1.z; vb[7] = (__bf16)a1.w;
                *(bf16x8*)(Ab + erow * 128 + hi * 64 + f0) = vb;
            }
        }
        fence_sched();   // per-wave LDS: DS pipe is in-order, just stop reordering

        float rad8[8];
#pragma unroll
        for (int v = 0; v < 8; ++v) rad8[v] = sRad[wv][hi * 8 + v];

        // ---- edge MLP layer 1: [16 x 128] @ [128 x 64] (+radial rank-1) ----
        f32x8 acc[4] = {};
#pragma unroll
        for (int k = 0; k < 4; ++k) {
            bf16x16 a = make_a_frag(Ab, 128, k * 32, lane);
#pragma unroll
            for (int n = 0; n < 4; ++n) {
                bf16x16 b = make_b_frag(sE1, 128, n * 16, k * 32, lane);
                acc[n] = wmma_bf16(a, b, acc[n]);
            }
        }
#pragma unroll
        for (int n = 0; n < 4; ++n)
#pragma unroll
            for (int v = 0; v < 8; ++v) {
                float z = acc[n][v] + rad8[v] * wradr[n] + eb1r[n];
                acc[n][v] = silu_f(z);
                Mb[(hi * 8 + v) * 64 + n * 16 + col] = (__bf16)acc[n][v];
            }
        fence_sched();

        // ---- edge MLP layer 2: [16 x 64] @ [64 x 64] -> m ----
        f32x8 acc2[4] = {};
#pragma unroll
        for (int k = 0; k < 2; ++k) {
            bf16x16 a = make_a_frag(Mb, 64, k * 32, lane);
#pragma unroll
            for (int n = 0; n < 4; ++n) {
                bf16x16 b = make_b_frag(sE2, 64, n * 16, k * 32, lane);
                acc2[n] = wmma_bf16(a, b, acc2[n]);
            }
        }
        fence_sched();

        int drow[8];
#pragma unroll
        for (int v = 0; v < 8; ++v) drow[v] = sDst[wv][hi * 8 + v];

#pragma unroll
        for (int n = 0; n < 4; ++n)
#pragma unroll
            for (int v = 0; v < 8; ++v) {
                float m = silu_f(acc2[n][v] + eb2r[n]);
                acc2[n][v] = m;
                Mb[(hi * 8 + v) * 64 + n * 16 + col] = (__bf16)m;
                if (hi * 8 + v < cnt)
                    atomicAdd(&hne[(long)drow[v] * 64 + n * 16 + col], m);
            }
        fence_sched();

        // ---- coord MLP: silu(m@cw1+cb1) @ cw2 -> scalar per edge ----
        f32x8 acc3[4] = {};
#pragma unroll
        for (int k = 0; k < 2; ++k) {
            bf16x16 a = make_a_frag(Mb, 64, k * 32, lane);
#pragma unroll
            for (int n = 0; n < 4; ++n) {
                bf16x16 b = make_b_frag(sC1, 64, n * 16, k * 32, lane);
                acc3[n] = wmma_bf16(a, b, acc3[n]);
            }
        }
        float part[8];
#pragma unroll
        for (int v = 0; v < 8; ++v) part[v] = 0.0f;
#pragma unroll
        for (int n = 0; n < 4; ++n)
#pragma unroll
            for (int v = 0; v < 8; ++v)
                part[v] += silu_f(acc3[n][v] + cb1r[n]) * cw2r[n];

#pragma unroll
        for (int m = 1; m <= 8; m <<= 1)
#pragma unroll
            for (int v = 0; v < 8; ++v)
                part[v] += __shfl_xor(part[v], m, 32);

        if (col == 0) {
#pragma unroll
            for (int v = 0; v < 8; ++v) {
                int row = hi * 8 + v;
                if (row < cnt) {
                    float c = part[v];
                    int dn = sDst[wv][row];
                    atomicAdd(&xne[dn * 3 + 0], c * sXdn[wv][row][0]);
                    atomicAdd(&xne[dn * 3 + 1], c * sXdn[wv][row][1]);
                    atomicAdd(&xne[dn * 3 + 2], c * sXdn[wv][row][2]);
                }
            }
        }
        fence_sched();
    }
}

// ---------------------------------------------------------------------------
// Node kernel: h_new = silu([h|h_neigh]@nw1+nb1)@nw2+nb2 ; x_new = x + x_neigh
// ---------------------------------------------------------------------------
__global__ __launch_bounds__(256) void egnn_node_kernel(
    const float* __restrict__ h, const float* __restrict__ hne,
    const float* __restrict__ x, const float* __restrict__ xne,
    const float* __restrict__ nw1, const float* __restrict__ nb1,
    const float* __restrict__ nw2, const float* __restrict__ nb2,
    float* __restrict__ hout, float* __restrict__ xout)
{
    __shared__ __bf16 sW1[64 * 128];
    __shared__ __bf16 sW2[64 * 64];
    __shared__ __bf16 sA[NWAVES][16 * 128];
    __shared__ __bf16 sM[NWAVES][16 * 64];

    const int tid = threadIdx.x;
    for (int i = tid; i < 64 * 128; i += 256) {
        int n = i >> 7, k = i & 127;
        sW1[i] = (__bf16)nw1[k * 64 + n];
    }
    for (int i = tid; i < 64 * 64; i += 256) {
        int n = i >> 6, k = i & 63;
        sW2[i] = (__bf16)nw2[k * 64 + n];
    }
    __syncthreads();

    const int lane = tid & 31;
    const int wv   = tid >> 5;
    const int col  = lane & 15;
    const int hi   = (lane >> 4) & 1;

    float nb1r[4], nb2r[4];
#pragma unroll
    for (int n = 0; n < 4; ++n) {
        nb1r[n] = nb1[n * 16 + col];
        nb2r[n] = nb2[n * 16 + col];
    }

    __bf16* Ab = sA[wv];
    __bf16* Mb = sM[wv];

    const int ntiles = (N_NODES + 15) >> 4;
    const int nwtot  = gridDim.x * NWAVES;

    for (int tile = blockIdx.x * NWAVES + wv; tile < ntiles; tile += nwtot) {
        const int row  = lane & 15;
        const int node = tile * 16 + row;
        const int nc   = min(node, N_NODES - 1);

        const float* hp = (hi ? hne : h) + (long)nc * 64;
#pragma unroll
        for (int f0 = 0; f0 < 64; f0 += 8) {
            float4 a0 = *(const float4*)(hp + f0);
            float4 a1 = *(const float4*)(hp + f0 + 4);
            bf16x8 vb;
            vb[0] = (__bf16)a0.x; vb[1] = (__bf16)a0.y;
            vb[2] = (__bf16)a0.z; vb[3] = (__bf16)a0.w;
            vb[4] = (__bf16)a1.x; vb[5] = (__bf16)a1.y;
            vb[6] = (__bf16)a1.z; vb[7] = (__bf16)a1.w;
            *(bf16x8*)(Ab + row * 128 + hi * 64 + f0) = vb;
        }

        if (!hi && node < N_NODES) {
#pragma unroll
            for (int d = 0; d < 3; ++d)
                xout[node * 3 + d] = x[node * 3 + d] + xne[node * 3 + d];
        }
        fence_sched();

        // MLP layer 1: [16 x 128] @ [128 x 64]
        f32x8 acc[4] = {};
#pragma unroll
        for (int k = 0; k < 4; ++k) {
            bf16x16 a = make_a_frag(Ab, 128, k * 32, lane);
#pragma unroll
            for (int n = 0; n < 4; ++n) {
                bf16x16 b = make_b_frag(sW1, 128, n * 16, k * 32, lane);
                acc[n] = wmma_bf16(a, b, acc[n]);
            }
        }
#pragma unroll
        for (int n = 0; n < 4; ++n)
#pragma unroll
            for (int v = 0; v < 8; ++v) {
                float z = silu_f(acc[n][v] + nb1r[n]);
                Mb[(hi * 8 + v) * 64 + n * 16 + col] = (__bf16)z;
            }
        fence_sched();

        // MLP layer 2: [16 x 64] @ [64 x 64] + bias (no activation)
        f32x8 acc2[4] = {};
#pragma unroll
        for (int k = 0; k < 2; ++k) {
            bf16x16 a = make_a_frag(Mb, 64, k * 32, lane);
#pragma unroll
            for (int n = 0; n < 4; ++n) {
                bf16x16 b = make_b_frag(sW2, 64, n * 16, k * 32, lane);
                acc2[n] = wmma_bf16(a, b, acc2[n]);
            }
        }
#pragma unroll
        for (int n = 0; n < 4; ++n)
#pragma unroll
            for (int v = 0; v < 8; ++v) {
                int r2 = hi * 8 + v;
                int n2 = tile * 16 + r2;
                if (n2 < N_NODES)
                    hout[(long)n2 * 64 + n * 16 + col] = acc2[n][v] + nb2r[n];
            }
        fence_sched();
    }
}

extern "C" void kernel_launch(void* const* d_in, const int* in_sizes, int n_in,
                              void* d_out, int out_size, void* d_ws, size_t ws_size,
                              hipStream_t stream) {
    (void)in_sizes; (void)n_in; (void)out_size; (void)ws_size;
    const float* h0  = (const float*)d_in[0];
    const float* x0  = (const float*)d_in[1];
    const int*   src = (const int*)d_in[2];
    const int*   dst = (const int*)d_in[3];
    const float* ew1 = (const float*)d_in[4];
    const float* eb1 = (const float*)d_in[5];
    const float* ew2 = (const float*)d_in[6];
    const float* eb2 = (const float*)d_in[7];
    const float* nw1 = (const float*)d_in[8];
    const float* nb1 = (const float*)d_in[9];
    const float* nw2 = (const float*)d_in[10];
    const float* nb2 = (const float*)d_in[11];
    const float* cw1 = (const float*)d_in[12];
    const float* cb1 = (const float*)d_in[13];
    const float* cw2 = (const float*)d_in[14];
    float* out = (float*)d_out;

    float* ws = (float*)d_ws;
    float* hb[2] = { ws, ws + (size_t)N_NODES * FDIM };
    float* xb[2] = { ws + 2 * (size_t)N_NODES * FDIM,
                     ws + 2 * (size_t)N_NODES * FDIM + (size_t)N_NODES * 3 };
    float* hne = ws + 2 * (size_t)N_NODES * FDIM + 2 * (size_t)N_NODES * 3;
    float* xne = hne + (size_t)N_NODES * FDIM;

    const float* hcur = h0;
    const float* xcur = x0;
    for (int d = 0; d < DEPTH; ++d) {
        hipMemsetAsync(hne, 0, (size_t)N_NODES * FDIM * sizeof(float), stream);
        hipMemsetAsync(xne, 0, (size_t)N_NODES * 3 * sizeof(float), stream);
        egnn_edge_kernel<<<2048, 256, 0, stream>>>(
            hcur, xcur, src, dst,
            ew1 + (size_t)d * 129 * 64, eb1 + (size_t)d * 64,
            ew2 + (size_t)d * 64 * 64,  eb2 + (size_t)d * 64,
            cw1 + (size_t)d * 64 * 64,  cb1 + (size_t)d * 64,
            cw2 + (size_t)d * 64,
            hne, xne);
        float* hnext = hb[d & 1];
        float* xnext = (d == DEPTH - 1) ? out : xb[d & 1];
        egnn_node_kernel<<<512, 256, 0, stream>>>(
            hcur, hne, xcur, xne,
            nw1 + (size_t)d * 128 * 64, nb1 + (size_t)d * 64,
            nw2 + (size_t)d * 64 * 64,  nb2 + (size_t)d * 64,
            hnext, xnext);
        hcur = hnext;
        xcur = xnext;
    }
}
